// VAE_Gumbel_50818053046916
// MI455X (gfx1250) — compile-verified, matrix operand order checked by reading, not run on hardware
//
#include <hip/hip_runtime.h>
#include <hip/hip_bf16.h>

// ---------------------------------------------------------------------------
// VAE-Gumbel forward for MI455X (gfx1250, wave32).
//  * all matmuls: v_wmma_f32_16x16x32_bf16, f32 accumulate
//  * GEMM operands pre-converted to bf16; weights pre-transposed to [N][K]
//  * A/B tiles DMA'd into LDS by the Tensor Data Mover (TENSORcnt),
//    double-buffered; TDM zero-fills OOB so the inner loop is branchless
//  * 50-step continuous top-k fused into one kernel (row in registers)
// ---------------------------------------------------------------------------

typedef __attribute__((ext_vector_type(16))) __bf16 v16bf;
typedef __attribute__((ext_vector_type(8)))  __bf16 v8bf;
typedef __attribute__((ext_vector_type(8)))  float  v8f;
typedef unsigned int v4u __attribute__((ext_vector_type(4)));
typedef int          v8i __attribute__((ext_vector_type(8)));
typedef int          v4i __attribute__((ext_vector_type(4)));

#define ACT_NONE 0
#define ACT_LEAKY 1
#define ACT_SIGMOID 2

// ---------------------------------------------------------------------------
// Tensor Data Mover: 2D bf16 tile (tile_d1 rows x tile_d0 cols) from a
// row-major tensor (row length = stride0 elements) into LDS at lds_off.
// tensor_dim0/1 are the remaining extents from the tile start so the TDM
// zero-fills out-of-range reads (handles K and N remainders for free).
// ---------------------------------------------------------------------------
__device__ __forceinline__ void tdm_load_2d(const __bf16* gptr, unsigned lds_off,
                                            unsigned dim0_len, unsigned dim1_len,
                                            unsigned long long stride0,
                                            unsigned tile_d0, unsigned tile_d1) {
  unsigned long long ga = (unsigned long long)(uintptr_t)gptr;
  v4u g0 = { 1u,                                         // count=1, user mode
             lds_off,                                    // LDS byte address
             (unsigned)(ga & 0xFFFFFFFFu),               // global_addr[31:0]
             (unsigned)((ga >> 32) & 0x01FFFFFFu) | (2u << 30) };  // [56:32] | type=2
  v8i g1 = { (int)(1u << 16),                            // data_size=1 (2 bytes)
             (int)((dim0_len & 0xFFFFu) << 16),          // tensor_dim0[15:0]
             (int)((dim0_len >> 16) | ((dim1_len & 0xFFFFu) << 16)),
             (int)((dim1_len >> 16) | (tile_d0 << 16)),  // tile_dim0
             (int)(tile_d1),                             // tile_dim1 (tile_dim2=0)
             (int)(unsigned)(stride0 & 0xFFFFFFFFu),     // tensor_dim0_stride[31:0]
             (int)(unsigned)((stride0 >> 32) & 0xFFFFu), // stride[47:32]
             0 };
  v4i gz = {0, 0, 0, 0};
#if __has_include(<hip/amd_detail/amd_gfx1250_TDM.h>)
  v8i gz8 = {0, 0, 0, 0, 0, 0, 0, 0};
  __builtin_amdgcn_tensor_load_to_lds(g0, g1, gz, gz, gz8, 0);
#else
  __builtin_amdgcn_tensor_load_to_lds(g0, g1, gz, gz, 0);
#endif
}

// ---------------------------------------------------------------------------
// GEMM: C[M,N] = act(A[M,K](bf16) @ Wt[N,K](bf16, pre-transposed) + bias[N])
// 256 threads = 8 waves (2 M x 4 N); WG tile 64x128; per-wave 32x32 (4 WMMA).
// TDM double-buffered LDS staging; branchless inner loop.
// ---------------------------------------------------------------------------
template <int ACT, bool OUTBF16>
__global__ __launch_bounds__(256) void gemm_tdm(
    const __bf16* __restrict__ A, const __bf16* __restrict__ Wt,
    const float* __restrict__ bias, float* __restrict__ Cf,
    __bf16* __restrict__ Cb, int M, int K, int N) {
  __shared__ __bf16 lA[2][64][32];   // [buf][m][k]
  __shared__ __bf16 lB[2][128][32];  // [buf][n][k]

  const int tid   = threadIdx.x;
  const int lane  = tid & 31;
  const int wave  = tid >> 5;
  const int waveM = wave >> 2;  // 0..1
  const int waveN = wave & 3;   // 0..3
  const int mBase = blockIdx.y * 64;
  const int nBase = blockIdx.x * 128;
  const int nT    = (K + 31) >> 5;

  const unsigned offA[2] = { (unsigned)(uintptr_t)&lA[0][0][0],
                             (unsigned)(uintptr_t)&lA[1][0][0] };
  const unsigned offB[2] = { (unsigned)(uintptr_t)&lB[0][0][0],
                             (unsigned)(uintptr_t)&lB[1][0][0] };

  v8f acc[2][2];
#pragma unroll
  for (int i = 0; i < 2; ++i)
#pragma unroll
    for (int j = 0; j < 2; ++j)
      acc[i][j] = (v8f){0.f, 0.f, 0.f, 0.f, 0.f, 0.f, 0.f, 0.f};

  // prologue: DMA tile 0 into buffer 0
  if (wave == 0) {
    tdm_load_2d(A  + (size_t)mBase * K, offA[0], (unsigned)K, (unsigned)(M - mBase),
                (unsigned long long)K, 32u, 64u);
    tdm_load_2d(Wt + (size_t)nBase * K, offB[0], (unsigned)K, (unsigned)(N - nBase),
                (unsigned long long)K, 32u, 128u);
    __builtin_amdgcn_s_wait_tensorcnt(0);
  }
  __syncthreads();

  const int mrow = lane & 15;
  const int kg   = lane >> 4;
  union V16 { v16bf v; v8bf h[2]; };

  for (int t = 0; t < nT; ++t) {
    const int cur = t & 1;
    // prefetch next tile into the other buffer while computing this one
    if (wave == 0 && (t + 1) < nT) {
      const int k1 = (t + 1) * 32;
      tdm_load_2d(A  + (size_t)mBase * K + k1, offA[cur ^ 1],
                  (unsigned)(K - k1), (unsigned)(M - mBase),
                  (unsigned long long)K, 32u, 64u);
      tdm_load_2d(Wt + (size_t)nBase * K + k1, offB[cur ^ 1],
                  (unsigned)(K - k1), (unsigned)(N - nBase),
                  (unsigned long long)K, 32u, 128u);
    }

    V16 afrag[2], bfrag[2];
#pragma unroll
    for (int fi = 0; fi < 2; ++fi) {
      const int ar = waveM * 32 + fi * 16 + mrow;
      afrag[fi].h[0] = *(const v8bf*)&lA[cur][ar][kg * 8];
      afrag[fi].h[1] = *(const v8bf*)&lA[cur][ar][16 + kg * 8];
    }
#pragma unroll
    for (int fj = 0; fj < 2; ++fj) {
      const int bc = waveN * 32 + fj * 16 + mrow;
      bfrag[fj].h[0] = *(const v8bf*)&lB[cur][bc][kg * 8];
      bfrag[fj].h[1] = *(const v8bf*)&lB[cur][bc][16 + kg * 8];
    }
#pragma unroll
    for (int fi = 0; fi < 2; ++fi)
#pragma unroll
      for (int fj = 0; fj < 2; ++fj)
        acc[fi][fj] = __builtin_amdgcn_wmma_f32_16x16x32_bf16(
            false, afrag[fi].v, false, bfrag[fj].v,
            (short)0, acc[fi][fj], false, false);

    if (wave == 0) __builtin_amdgcn_s_wait_tensorcnt(0);
    __syncthreads();
  }

  // epilogue: bias + activation + store (f32 or bf16)
  const int nlane = lane & 15;
  const int mhalf = (lane >> 4) * 8;
#pragma unroll
  for (int fj = 0; fj < 2; ++fj) {
    const int n = nBase + waveN * 32 + fj * 16 + nlane;
    if (n >= N) continue;
    const float bs = bias[n];
#pragma unroll
    for (int fi = 0; fi < 2; ++fi) {
#pragma unroll
      for (int r = 0; r < 8; ++r) {
        const int m = mBase + waveM * 32 + fi * 16 + mhalf + r;
        if (m < M) {
          float v = acc[fi][fj][r] + bs;
          if (ACT == ACT_LEAKY)   v = (v > 0.f) ? v : 0.01f * v;
          if (ACT == ACT_SIGMOID) v = 1.f / (1.f + __expf(-v));
          if (OUTBF16) Cb[(size_t)m * N + n] = (__bf16)v;
          else         Cf[(size_t)m * N + n] = v;
        }
      }
    }
  }
}

// ---------------------------------------------------------------------------
// Block-wide reductions (wave32 shuffles + LDS across waves)
// ---------------------------------------------------------------------------
__device__ __forceinline__ float wredSum(float v) {
#pragma unroll
  for (int o = 16; o > 0; o >>= 1) v += __shfl_xor(v, o, 32);
  return v;
}
__device__ __forceinline__ float wredMax(float v) {
#pragma unroll
  for (int o = 16; o > 0; o >>= 1) v = fmaxf(v, __shfl_xor(v, o, 32));
  return v;
}
__device__ float bredSum(float v, float* sm) {
  int tid = threadIdx.x, lane = tid & 31, wid = tid >> 5;
  int nw = (blockDim.x + 31) >> 5;
  v = wredSum(v);
  if (lane == 0) sm[wid] = v;
  __syncthreads();
  if (wid == 0) {
    float t = (lane < nw) ? sm[lane] : 0.f;
    t = wredSum(t);
    if (lane == 0) sm[0] = t;
  }
  __syncthreads();
  float r = sm[0];
  __syncthreads();
  return r;
}
__device__ float bredMax(float v, float* sm) {
  int tid = threadIdx.x, lane = tid & 31, wid = tid >> 5;
  int nw = (blockDim.x + 31) >> 5;
  v = wredMax(v);
  if (lane == 0) sm[wid] = v;
  __syncthreads();
  if (wid == 0) {
    float t = (lane < nw) ? sm[lane] : -3.0e38f;
    t = wredMax(t);
    if (lane == 0) sm[0] = t;
  }
  __syncthreads();
  float r = sm[0];
  __syncthreads();
  return r;
}

// ---------------------------------------------------------------------------
// Fused gumbel noise + 50-step continuous top-k + mask multiply (bf16 out).
// One workgroup per row; whole 5000-wide row resident in registers.
// ---------------------------------------------------------------------------
#define TK_THREADS 512
#define TK_SLOTS 10
#define EPSILON_C 1e-30f
#define TOPK_K 50
#define INV_T 10.0f

__global__ __launch_bounds__(TK_THREADS) void topk_fused(
    const float* __restrict__ x, const float* __restrict__ wlog,
    const float* __restrict__ u, __bf16* __restrict__ xm, int D) {
  __shared__ float sm[32];
  const int row = blockIdx.x;
  const int tid = threadIdx.x;
  const size_t base = (size_t)row * D;

  float wv[TK_SLOTS], kh[TK_SLOTS], oh[TK_SLOTS];
#pragma unroll
  for (int j = 0; j < TK_SLOTS; ++j) {
    int i = tid + j * TK_THREADS;
    kh[j] = 0.f; oh[j] = 0.f;
    if (i < D) {
      float uu = (1.0f - EPSILON_C) * u[base + i] + EPSILON_C;
      wv[j] = wlog[base + i] + __logf(-__logf(uu));
    } else {
      wv[j] = -3.0e38f;
    }
  }

  for (int it = 0; it < TOPK_K; ++it) {
    float m = -3.0e38f;
#pragma unroll
    for (int j = 0; j < TK_SLOTS; ++j) {
      wv[j] += __logf(fmaxf(1.0f - oh[j], EPSILON_C));
      m = fmaxf(m, wv[j]);
    }
    m = bredMax(m, sm);
    float s = 0.f;
#pragma unroll
    for (int j = 0; j < TK_SLOTS; ++j) {
      float e = __expf((wv[j] - m) * INV_T);
      oh[j] = e;
      s += e;
    }
    s = bredSum(s, sm);
    float inv = 1.0f / s;
#pragma unroll
    for (int j = 0; j < TK_SLOTS; ++j) {
      oh[j] *= inv;
      kh[j] += oh[j];
    }
  }

#pragma unroll
  for (int j = 0; j < TK_SLOTS; ++j) {
    int i = tid + j * TK_THREADS;
    if (i < D) xm[base + i] = (__bf16)(x[base + i] * kh[j]);
  }
}

// ---------------------------------------------------------------------------
// Elementwise / pre-pass kernels
// ---------------------------------------------------------------------------
__global__ __launch_bounds__(256) void cvt_bf16(const float* __restrict__ in,
                                                __bf16* __restrict__ out, size_t n) {
  size_t i = blockIdx.x * (size_t)blockDim.x + threadIdx.x;
  if (i < n) out[i] = (__bf16)in[i];
}

// W[K][N] f32 -> Wt[N][K] bf16 (LDS-tiled transpose, coalesced both sides)
__global__ __launch_bounds__(256) void transpose_bf16(
    const float* __restrict__ W, __bf16* __restrict__ Wt, int K, int N) {
  __shared__ float t[32][33];
  const int kb = blockIdx.y * 32, nb = blockIdx.x * 32;
  const int lx = threadIdx.x & 31, ly = threadIdx.x >> 5;  // 8 rows per pass
#pragma unroll
  for (int r = ly; r < 32; r += 8) {
    int k = kb + r, n = nb + lx;
    t[r][lx] = (k < K && n < N) ? W[(size_t)k * N + n] : 0.f;
  }
  __syncthreads();
#pragma unroll
  for (int r = ly; r < 32; r += 8) {
    int n = nb + r, k = kb + lx;
    if (n < N && k < K) Wt[(size_t)n * K + k] = (__bf16)t[lx][r];
  }
}

__global__ __launch_bounds__(256) void bn_stats(
    const float* __restrict__ h, float* __restrict__ stats, int M, int Hc) {
  __shared__ float sm[32];
  const int c = blockIdx.x;
  float s = 0.f, ss = 0.f;
  for (int r = threadIdx.x; r < M; r += blockDim.x) {
    float v = h[(size_t)r * Hc + c];
    s += v; ss += v * v;
  }
  s  = bredSum(s, sm);
  ss = bredSum(ss, sm);
  if (threadIdx.x == 0) {
    float mu = s / (float)M;
    stats[c]      = mu;
    stats[Hc + c] = ss / (float)M - mu * mu;
  }
}

__global__ __launch_bounds__(256) void bn_relu_bf16(
    const float* __restrict__ h, const float* __restrict__ stats,
    const float* __restrict__ g, const float* __restrict__ b,
    __bf16* __restrict__ out, int tot, int Hc) {
  int idx = blockIdx.x * blockDim.x + threadIdx.x;
  if (idx >= tot) return;
  int c = idx % Hc;
  float v = (h[idx] - stats[c]) * rsqrtf(stats[Hc + c] + 1e-5f) * g[c] + b[c];
  out[idx] = (__bf16)fmaxf(v, 0.f);
}

__global__ __launch_bounds__(256) void reparam_bf16(
    const float* __restrict__ mu, const float* __restrict__ lv,
    const float* __restrict__ eps, __bf16* __restrict__ z, int tot) {
  int idx = blockIdx.x * blockDim.x + threadIdx.x;
  if (idx >= tot) return;
  z[idx] = (__bf16)(mu[idx] + eps[idx] * __expf(0.5f * lv[idx]));
}

// ---------------------------------------------------------------------------
extern "C" void kernel_launch(void* const* d_in, const int* in_sizes, int n_in,
                              void* d_out, int out_size, void* d_ws, size_t ws_size,
                              hipStream_t stream) {
  const int B = 4096, D = 5000, H = 512, Z = 64;

  const float* x       = (const float*)d_in[0];
  const float* noise_u = (const float*)d_in[1];
  const float* eps     = (const float*)d_in[2];
  const float* wc_w1   = (const float*)d_in[3];
  const float* wc_b1   = (const float*)d_in[4];
  const float* bn_g    = (const float*)d_in[5];
  const float* bn_b    = (const float*)d_in[6];
  const float* wc_w2   = (const float*)d_in[7];
  const float* wc_b2   = (const float*)d_in[8];
  const float* enc_w1  = (const float*)d_in[9];
  const float* enc_b1  = (const float*)d_in[10];
  const float* enc_w2  = (const float*)d_in[11];
  const float* enc_b2  = (const float*)d_in[12];
  const float* enc_w3  = (const float*)d_in[13];
  const float* enc_b3  = (const float*)d_in[14];
  const float* enc_w4  = (const float*)d_in[15];
  const float* enc_b4  = (const float*)d_in[16];
  const float* mean_w  = (const float*)d_in[17];
  const float* mean_b  = (const float*)d_in[18];
  const float* lv_w    = (const float*)d_in[19];
  const float* lv_b    = (const float*)d_in[20];
  const float* dec_w1  = (const float*)d_in[21];
  const float* dec_b1  = (const float*)d_in[22];
  const float* dec_w2  = (const float*)d_in[23];
  const float* dec_b2  = (const float*)d_in[24];

  float* out_mu_x = (float*)d_out;             // [B,D]
  float* out_mu   = out_mu_x + (size_t)B * D;  // [B,Z]
  float* out_lv   = out_mu + (size_t)B * Z;    // [B,Z]

  // workspace carve
  char* p = (char*)d_ws;
  auto alloc = [&](size_t bytes) -> void* {
    void* r = (void*)p;
    p += (bytes + 255) & ~(size_t)255;
    return r;
  };
  float*  wbuf  = (float*) alloc((size_t)B * D * 4);      // gumbel keys (f32)
  float*  hbuf  = (float*) alloc((size_t)B * H * 4);      // pre-BN hidden (f32)
  float*  stat  = (float*) alloc((size_t)2 * H * 4);
  __bf16* xbf   = (__bf16*)alloc((size_t)B * D * 2);
  __bf16* xmbf  = (__bf16*)alloc((size_t)B * D * 2);
  __bf16* hrel  = (__bf16*)alloc((size_t)B * H * 2);
  __bf16* actA  = (__bf16*)alloc((size_t)B * 2 * H * 2);
  __bf16* actB  = (__bf16*)alloc((size_t)B * 2 * H * 2);
  __bf16* zbf   = (__bf16*)alloc((size_t)B * Z * 2);
  __bf16* t_wc1 = (__bf16*)alloc((size_t)D * H * 2);      // [H][D]
  __bf16* t_wc2 = (__bf16*)alloc((size_t)H * D * 2);      // [D][H]
  __bf16* t_e1  = (__bf16*)alloc((size_t)D * 2 * H * 2);  // [2H][D]
  __bf16* t_e2  = (__bf16*)alloc((size_t)2 * H * H * 2);  // [H][2H]
  __bf16* t_e3  = (__bf16*)alloc((size_t)H * H * 2);      // [H][H]
  __bf16* t_e4  = (__bf16*)alloc((size_t)H * H * 2);      // [H][H]
  __bf16* t_mw  = (__bf16*)alloc((size_t)H * Z * 2);      // [Z][H]
  __bf16* t_lw  = (__bf16*)alloc((size_t)H * Z * 2);      // [Z][H]
  __bf16* t_d1  = (__bf16*)alloc((size_t)Z * 2 * H * 2);  // [2H][Z]
  __bf16* t_d2  = (__bf16*)alloc((size_t)2 * H * D * 2);  // [D][2H]

  const dim3 blk(256);
  auto gg = [](int M, int N) { return dim3((N + 127) / 128, (M + 63) / 64); };
  auto tg = [](int K, int N) { return dim3((N + 31) / 32, (K + 31) / 32); };

  // ---- pre-pass: operand precision conversion ----
  cvt_bf16<<<dim3((unsigned)(((size_t)B * D + 255) / 256)), blk, 0, stream>>>(x, xbf, (size_t)B * D);
  transpose_bf16<<<tg(D, H),     blk, 0, stream>>>(wc_w1,  t_wc1, D, H);
  transpose_bf16<<<tg(H, D),     blk, 0, stream>>>(wc_w2,  t_wc2, H, D);
  transpose_bf16<<<tg(D, 2 * H), blk, 0, stream>>>(enc_w1, t_e1,  D, 2 * H);
  transpose_bf16<<<tg(2 * H, H), blk, 0, stream>>>(enc_w2, t_e2,  2 * H, H);
  transpose_bf16<<<tg(H, H),     blk, 0, stream>>>(enc_w3, t_e3,  H, H);
  transpose_bf16<<<tg(H, H),     blk, 0, stream>>>(enc_w4, t_e4,  H, H);
  transpose_bf16<<<tg(H, Z),     blk, 0, stream>>>(mean_w, t_mw,  H, Z);
  transpose_bf16<<<tg(H, Z),     blk, 0, stream>>>(lv_w,   t_lw,  H, Z);
  transpose_bf16<<<tg(Z, 2 * H), blk, 0, stream>>>(dec_w1, t_d1,  Z, 2 * H);
  transpose_bf16<<<tg(2 * H, D), blk, 0, stream>>>(dec_w2, t_d2,  2 * H, D);

  // ---- weight_creator: h = x @ wc_w1 + b -> BN(train) -> ReLU ----
  gemm_tdm<ACT_NONE, false><<<gg(B, H), blk, 0, stream>>>(
      xbf, t_wc1, wc_b1, hbuf, nullptr, B, D, H);
  bn_stats<<<dim3(H), blk, 0, stream>>>(hbuf, stat, B, H);
  bn_relu_bf16<<<dim3((B * H + 255) / 256), blk, 0, stream>>>(
      hbuf, stat, bn_g, bn_b, hrel, B * H, H);

  // ---- w = h @ wc_w2 + b (f32, feeds top-k) ----
  gemm_tdm<ACT_NONE, false><<<gg(B, D), blk, 0, stream>>>(
      hrel, t_wc2, wc_b2, wbuf, nullptr, B, H, D);

  // ---- fused gumbel + continuous top-k + mask multiply ----
  topk_fused<<<dim3(B), dim3(TK_THREADS), 0, stream>>>(x, wbuf, noise_u, xmbf, D);

  // ---- encoder ----
  gemm_tdm<ACT_LEAKY, true><<<gg(B, 2 * H), blk, 0, stream>>>(
      xmbf, t_e1, enc_b1, nullptr, actA, B, D, 2 * H);
  gemm_tdm<ACT_LEAKY, true><<<gg(B, H), blk, 0, stream>>>(
      actA, t_e2, enc_b2, nullptr, actB, B, 2 * H, H);
  gemm_tdm<ACT_LEAKY, true><<<gg(B, H), blk, 0, stream>>>(
      actB, t_e3, enc_b3, nullptr, actA, B, H, H);
  gemm_tdm<ACT_LEAKY, true><<<gg(B, H), blk, 0, stream>>>(
      actA, t_e4, enc_b4, nullptr, actB, B, H, H);

  // ---- latent heads (f32, straight into d_out) ----
  gemm_tdm<ACT_NONE, false><<<gg(B, Z), blk, 0, stream>>>(
      actB, t_mw, mean_b, out_mu, nullptr, B, H, Z);
  gemm_tdm<ACT_NONE, false><<<gg(B, Z), blk, 0, stream>>>(
      actB, t_lw, lv_b, out_lv, nullptr, B, H, Z);

  // ---- reparameterize ----
  reparam_bf16<<<dim3((B * Z + 255) / 256), blk, 0, stream>>>(
      out_mu, out_lv, eps, zbf, B * Z);

  // ---- decoder ----
  gemm_tdm<ACT_LEAKY, true><<<gg(B, 2 * H), blk, 0, stream>>>(
      zbf, t_d1, dec_b1, nullptr, actA, B, Z, 2 * H);
  gemm_tdm<ACT_SIGMOID, false><<<gg(B, D), blk, 0, stream>>>(
      actA, t_d2, dec_b2, out_mu_x, nullptr, B, 2 * H, D);
}
